// GeometricEulerDecoder_51926154609242
// MI455X (gfx1250) — matrix operand order repaired and checked by reading, not compiled, not run on hardware
//
#include <hip/hip_runtime.h>
#include <hip/hip_bf16.h>
#include <math.h>

typedef __attribute__((ext_vector_type(16))) _Float16 v16h;
typedef __attribute__((ext_vector_type(8)))  float    v8f;

// ---------------- d_ws layout ----------------
// bytes [0, 12288): _Float16 Bmat[192][32]   (N-major, K contiguous)
//   cols  0.. 44 : T442  (n = J*5+K, rows I=0..8   -> f4)
//   cols 48..112 : T462  (n = 48+J*5+K, rows I=0..8 -> f4, J indexes f6)
//   cols 120..184: T662  (n = 120+J*5+K, rows 9+I   -> f6)
// bytes [12288, ...): float table, indices below
#define WSF_DIAG440 0     // 9
#define WSF_DIAG660 9     // 13
#define WSF_GWA     22    // 8  : C0*w1_44_s2[8+w]
#define WSF_GWB     30    // 8  : C0*w1_66_s2[8+w]
#define WSF_TWA     38    // 8  : C2*w1_44_t[w]
#define WSF_TWB     46    // 8  : C2*(w1_46_t[w]+w1_64_t[w])
#define WSF_TWC     54    // 8  : C2*w1_66_t[w]
#define WSF_W2TT    62    // 128: w2_tt[u][v][wo]
#define WSF_C221    190   // 75 : CT2 * C221[I][J][K]
#define WSF_CSIG    265   // 1
#define WSF_COUNT   266

// ============ setup: Wigner 3j (real basis) + weight folding ============
__device__ __forceinline__ double d_fact(int n){ double f=1.0; for(int i=2;i<=n;++i) f*=(double)i; return f; }
__device__ __forceinline__ int imax2(int a,int b){return a>b?a:b;}
__device__ __forceinline__ int imin2(int a,int b){return a<b?a:b;}

__device__ double su2_cg(int j1,int j2,int j3,int m1,int m2,int m3){
  if(m3 != m1+m2) return 0.0;
  int vmin = imax2(imax2(-j1+j2+m3, -j1+m1), 0);
  int vmax = imin2(imin2(j2+j3+m1, j3-j1+j2), j3+m3);
  if(vmax < vmin) return 0.0;
  double c = sqrt((double)(2*j3+1)*d_fact(j3+j1-j2)*d_fact(j3-j1+j2)*d_fact(j1+j2-j3)*d_fact(j3+m3)*d_fact(j3-m3)
                 /(d_fact(j1+j2+j3+1)*d_fact(j1-m1)*d_fact(j1+m1)*d_fact(j2-m2)*d_fact(j2+m2)));
  double s=0.0;
  for(int v=vmin;v<=vmax;++v){
    double sg = ((v+j2+m2)&1)? -1.0:1.0;
    s += sg*d_fact(j2+j3+m1-v)*d_fact(j1-m1+v)
         /(d_fact(v)*d_fact(j3-j1+j2-v)*d_fact(j3+m3-v)*d_fact(v+j1-j2-m3));
  }
  return c*s;
}

// real_basis(l)[row a][col i] including (-i)^l phase
__device__ void rb_entry(int l,int a,int i,double& re,double& im){
  const double inv = 0.70710678118654752440;
  double qr=0.0, qi=0.0;
  int m = a - l;
  if(m < 0){ if(i == l-m) qr = inv; else if(i == l+m) qi = -inv; }
  else if(m == 0){ if(i == l) qr = 1.0; }
  else { double s = (m&1)? -1.0:1.0; if(i == l+m) qr = s*inv; else if(i == l-m) qi = s*inv; }
  double pr, pi;
  switch(l & 3){ case 0: pr=1; pi=0; break; case 1: pr=0; pi=-1; break;
                 case 2: pr=-1; pi=0; break; default: pr=0; pi=1; }
  re = qr*pr - qi*pi; im = qr*pi + qi*pr;
}

// Re[ sum_{a,b,c} Q1[a][i] Q2[b][j] conj(Q3[k][c]) CG(a,b,c) ]
__device__ double w3j_entry(int l1,int l2,int l3,int i,int j,int k){
  double acc = 0.0;
  int as[2]={i,2*l1-i}, bs[2]={j,2*l2-j}, cs[2]={k,2*l3-k};
  int na=(as[0]==as[1])?1:2, nb=(bs[0]==bs[1])?1:2, nc=(cs[0]==cs[1])?1:2;
  for(int ai=0; ai<na; ++ai){
    double q1r,q1i; rb_entry(l1,as[ai],i,q1r,q1i);
    if(q1r==0.0 && q1i==0.0) continue;
    for(int bi=0; bi<nb; ++bi){
      double q2r,q2i; rb_entry(l2,bs[bi],j,q2r,q2i);
      if(q2r==0.0 && q2i==0.0) continue;
      double t12r=q1r*q2r-q1i*q2i, t12i=q1r*q2i+q1i*q2r;
      for(int ci=0; ci<nc; ++ci){
        double q3r,q3i; rb_entry(l3,k,cs[ci],q3r,q3i);  // Q3[k][c]
        q3i = -q3i;                                     // conj
        if(q3r==0.0 && q3i==0.0) continue;
        double cg = su2_cg(l1,l2,l3, as[ai]-l1, bs[bi]-l2, cs[ci]-l3);
        if(cg==0.0) continue;
        acc += (t12r*q3r - t12i*q3i)*cg;
      }
    }
  }
  return acc;
}

__global__ __launch_bounds__(128)
void ged_setup_kernel(const float* __restrict__ w44s2, const float* __restrict__ w66s2,
                      const float* __restrict__ w44t,  const float* __restrict__ w46t,
                      const float* __restrict__ w64t,  const float* __restrict__ w66t,
                      const float* __restrict__ w2tt,  void* __restrict__ ws){
  __shared__ double dbuf[845];
  _Float16* bm  = (_Float16*)ws;
  float*    wsf = (float*)((char*)ws + 12288);
  const int tid = threadIdx.x;
  for(int i=tid;i<192*32;i+=blockDim.x) bm[i] = (_Float16)0.0f;
  __syncthreads();

  const int L1[6]={4,4,6,4,6,2}, L2[6]={4,6,6,4,6,2}, L3[6]={2,2,2,0,0,1};
  for(int t=0;t<6;++t){
    int l1=L1[t], l2=L2[t], l3=L3[t];
    int n1=2*l1+1, n2=2*l2+1, n3=2*l3+1, sz=n1*n2*n3;
    for(int idx=tid; idx<sz; idx+=blockDim.x){
      int k=idx%n3, j=(idx/n3)%n2, i=idx/(n3*n2);
      dbuf[idx] = w3j_entry(l1,l2,l3,i,j,k);
    }
    __syncthreads();
    if(tid==0){
      double s=0.0; for(int q=0;q<sz;++q) s += dbuf[q]*dbuf[q];
      double inv = 1.0/sqrt(s);
      if(t==0){        // (4,4,2) -> Bmat cols 0..44, rows I(f4)
        for(int i=0;i<9;++i) for(int j=0;j<9;++j) for(int k=0;k<5;++k)
          bm[(j*5+k)*32 + i] = (_Float16)(float)(dbuf[(i*9+j)*5+k]*inv);
      } else if(t==1){ // (4,6,2) -> cols 48.., rows I(f4), J = f6
        for(int i=0;i<9;++i) for(int j=0;j<13;++j) for(int k=0;k<5;++k)
          bm[(48 + j*5+k)*32 + i] = (_Float16)(float)(dbuf[(i*13+j)*5+k]*inv);
      } else if(t==2){ // (6,6,2) -> cols 120.., rows 9+I(f6)
        for(int i=0;i<13;++i) for(int j=0;j<13;++j) for(int k=0;k<5;++k)
          bm[(120 + j*5+k)*32 + 9+i] = (_Float16)(float)(dbuf[(i*13+j)*5+k]*inv);
      } else if(t==3){ // (4,4,0) diagonal
        for(int i=0;i<9;++i)  wsf[WSF_DIAG440+i] = (float)(dbuf[(i*9+i)]*inv);
      } else if(t==4){ // (6,6,0) diagonal
        for(int i=0;i<13;++i) wsf[WSF_DIAG660+i] = (float)(dbuf[(i*13+i)]*inv);
      } else {         // (2,2,1), prefold CT2 = sqrt(3/512)
        const double CT2 = sqrt(3.0/512.0);
        for(int i=0;i<5;++i) for(int j=0;j<5;++j) for(int k=0;k<3;++k)
          wsf[WSF_C221 + (i*5+j)*3+k] = (float)(CT2*dbuf[(i*5+j)*3+k]*inv);
      }
    }
    __syncthreads();
  }

  if(tid==0){
    const double C0 = sqrt(0.5), C2 = sqrt(1.25);
    for(int w=0;w<8;++w){
      wsf[WSF_GWA+w] = (float)(C0*(double)w44s2[8+w]);
      wsf[WSF_GWB+w] = (float)(C0*(double)w66s2[8+w]);
      wsf[WSF_TWA+w] = (float)(C2*(double)w44t[w]);
      wsf[WSF_TWB+w] = (float)(C2*((double)w46t[w]+(double)w64t[w]));  // T642 == T462
      wsf[WSF_TWC+w] = (float)(C2*(double)w66t[w]);
    }
    for(int i=0;i<128;++i) wsf[WSF_W2TT+i] = w2tt[i];
    // csig = (E[sigmoid(x)^2])^{-1/2}, x~N(0,1); Simpson on [-12,12]
    const int N = 4096; const double a=-12.0, h=24.0/N;
    double s=0.0;
    for(int i=0;i<=N;++i){
      double x = a + h*i;
      double sg = 1.0/(1.0+exp(-x));
      double f  = exp(-0.5*x*x)*sg*sg;
      double wgt = (i==0||i==N)?1.0:((i&1)?4.0:2.0);
      s += wgt*f;
    }
    s *= h/3.0 * 0.39894228040143267794; // 1/sqrt(2*pi)
    wsf[WSF_CSIG] = (float)(1.0/sqrt(s));
  }
}

// ======================= main kernel =======================
// 64 threads (2 waves), 64 elements per block.
__global__ __launch_bounds__(64)
void ged_main_kernel(const float* __restrict__ feat, const void* __restrict__ ws,
                     float* __restrict__ out, int B){
  __shared__ float    sF [64*23];    // f32 features (padded stride 23)
  __shared__ _Float16 sFH[64*34];    // f16 features, K padded to 32 (stride 34)
  __shared__ _Float16 sB [192*34];   // Bmat [N][K] (stride 34)
  __shared__ _Float16 sY [64*196];   // GEMM result [elem][192] (stride 196)
  __shared__ float    sTab[WSF_COUNT];

  const int tid  = threadIdx.x;
  const int base = blockIdx.x * 64;
  const _Float16* bmat = (const _Float16*)ws;
  const float*    wsf  = (const float*)((const char*)ws + 12288);

  // ---- stage features + tables into LDS (coalesced) ----
  for(int i=tid; i<64*22; i+=64){
    int e = i/22, j = i - e*22;
    float v = (base + e < B) ? feat[(size_t)base*22 + i] : 0.0f;
    sF [e*23 + j] = v;
    sFH[e*34 + j] = (_Float16)v;
  }
  for(int i=tid; i<64*12; i+=64){ int e=i/12, j=22+(i-e*12); sFH[e*34+j] = (_Float16)0.0f; }
  for(int i=tid; i<192*32; i+=64){ sB[(i>>5)*34 + (i&31)] = bmat[i]; }
  for(int i=tid; i<WSF_COUNT; i+=64) sTab[i] = wsf[i];
  __syncthreads();

  // ---- WMMA: Y[elem, n] = sum_k f16(feat)[elem,k] * Bmat[k,n], K=32 one shot ----
  {
    const int lane = tid & 31, lhi = lane >> 4, lm = lane & 15;
    const int wv   = tid >> 5;
    v16h afr[2];
    #pragma unroll
    for(int m=0; m<2; ++m){
      int mt = wv*2 + m;
      int arow = (mt*16 + lm)*34;
      #pragma unroll
      for(int v=0; v<8; ++v){
        int kb = ((v&4)?16:0) + (lhi?8:0) + ((v&3)*2);   // 16-bit A 16x32 layout
        afr[m][2*v]   = sFH[arow + kb];
        afr[m][2*v+1] = sFH[arow + kb+1];
      }
    }
    #pragma unroll
    for(int nt=0; nt<12; ++nt){
      v16h bfr;
      int bcol = (nt*16 + lm)*34;
      int kb0  = lhi ? 8 : 0;                             // B: K striped, N = lane
      #pragma unroll
      for(int v=0; v<8; ++v){ int kb = ((v&4)?16:0) + kb0 + ((v&3)*2); bfr[2*v] = sB[bcol+kb]; bfr[2*v+1] = sB[bcol+kb+1]; }
      #pragma unroll
      for(int m=0; m<2; ++m){
        v8f c = {};
        c = __builtin_amdgcn_wmma_f32_16x16x32_f16(false, afr[m], false, bfr,
                                                   (short)0, c, false, false);
        int erow = (wv*2 + m)*16 + (lhi<<3);
        #pragma unroll
        for(int r=0; r<8; ++r) sY[(erow+r)*196 + nt*16 + lm] = (_Float16)c[r];
      }
    }
  }
  __syncthreads();

  if(base + tid >= B) return;

  // ---- per-thread scalar pipeline (element = tid) ----
  float f4[9], f6[13];
  {
    const float* fr = &sF[tid*23];
    #pragma unroll
    for(int i=0;i<9;++i)  f4[i] = fr[i];
    #pragma unroll
    for(int i=0;i<13;++i) f6[i] = fr[9+i];
  }
  float S440=0.f, S660=0.f;
  #pragma unroll
  for(int i=0;i<9;++i)  S440 += sTab[WSF_DIAG440+i]*f4[i]*f4[i];
  #pragma unroll
  for(int i=0;i<13;++i) S660 += sTab[WSF_DIAG660+i]*f6[i]*f6[i];

  // step 2 of bilinears: q_K = sum_J f_J * Y[(J,K)]
  float q442[5]={0,0,0,0,0}, q462[5]={0,0,0,0,0}, q662[5]={0,0,0,0,0};
  {
    const _Float16* yr = &sY[tid*196];
    #pragma unroll
    for(int J=0;J<9;++J){ float fj=f4[J];
      #pragma unroll
      for(int K=0;K<5;++K) q442[K] += fj*(float)yr[J*5+K];
    }
    #pragma unroll
    for(int J=0;J<13;++J){ float fj=f6[J];
      #pragma unroll
      for(int K=0;K<5;++K){ q462[K] += fj*(float)yr[48+J*5+K];
                            q662[K] += fj*(float)yr[120+J*5+K]; }
    }
  }

  // gates (only 8..15 survive: gv==0 exactly) and gated tensor gt
  const float csig = sTab[WSF_CSIG];
  float gt[8][5];
  #pragma unroll
  for(int w=0;w<8;++w){
    float s = sTab[WSF_GWA+w]*S440 + sTab[WSF_GWB+w]*S660;
    float g = csig / (1.f + __expf(-s));
    float ta=sTab[WSF_TWA+w], tb=sTab[WSF_TWB+w], tc=sTab[WSF_TWC+w];
    #pragma unroll
    for(int K=0;K<5;++K) gt[w][K] = g*(ta*q442[K] + tb*q462[K] + tc*q662[K]);
  }

  // layer 2, surviving tt path: G[I][J][wo] = sum_{u,v} w2tt[u,v,wo] gt_uI gt_vJ
  float G[5][5][2] = {};
  #pragma unroll
  for(int wo=0; wo<2; ++wo){
    for(int v=0; v<8; ++v){
      float h[5] = {0,0,0,0,0};
      #pragma unroll
      for(int u=0; u<8; ++u){
        float wv = sTab[WSF_W2TT + u*16 + v*2 + wo];
        #pragma unroll
        for(int I=0;I<5;++I) h[I] += wv*gt[u][I];
      }
      #pragma unroll
      for(int I=0;I<5;++I){
        float hi = h[I];
        #pragma unroll
        for(int J=0;J<5;++J) G[I][J][wo] += hi*gt[v][J];
      }
    }
  }
  float vec[2][3] = {};
  for(int I=0;I<5;++I)
    for(int J=0;J<5;++J){
      const float* c3 = &sTab[WSF_C221 + (I*5+J)*3];
      float g0=G[I][J][0], g1=G[I][J][1];
      #pragma unroll
      for(int K=0;K<3;++K){ float c=c3[K]; vec[0][K]+=c*g0; vec[1][K]+=c*g1; }
    }

  // Gram-Schmidt + Euler
  float vpx=vec[0][0], vpy=vec[0][1], vpz=vec[0][2];
  float vsx=vec[1][0], vsy=vec[1][1], vsz=vec[1][2];
  float n1 = sqrtf(vpx*vpx+vpy*vpy+vpz*vpz);
  float r1 = 1.f/fmaxf(n1, 1e-6f);
  float zx=vpx*r1, zy=vpy*r1, zz=vpz*r1;
  float dot = zx*vsx + zy*vsy + zz*vsz;
  float wx=vsx-dot*zx, wy=vsy-dot*zy, wz=vsz-dot*zz;
  float n2 = sqrtf(wx*wx+wy*wy+wz*wz);
  float r2 = 1.f/fmaxf(n2, 1e-6f);
  float xx=wx*r2, xy=wy*r2, xz=wz*r2;
  float yx = zy*xz - zz*xy;
  float yy = zz*xx - zx*xz;
  float yz = zx*xy - zy*xx;

  float czz  = fminf(1.f, fmaxf(-1.f, zz));
  float beta = acosf(czz);
  float sb   = sinf(beta);
  bool  safe = fabsf(sb) > 1e-6f;
  float alpha = safe ? atan2f(zy, zx) : 0.f;
  float gamma = safe ? atan2f(yz, -xz) : atan2f(-yx, yy);

  size_t b = (size_t)base + tid;
  out[b]              = alpha;
  out[(size_t)B + b]  = beta;
  out[(size_t)2*B + b]= gamma;
  float* R = out + (size_t)3*B + b*9;   // R[i][j], cols = (x,y,z)
  R[0]=xx; R[1]=yx; R[2]=zx;
  R[3]=xy; R[4]=yy; R[5]=zy;
  R[6]=xz; R[7]=yz; R[8]=zz;
}

extern "C" void kernel_launch(void* const* d_in, const int* in_sizes, int n_in,
                              void* d_out, int out_size, void* d_ws, size_t ws_size,
                              hipStream_t stream){
  const float* feat = (const float*)d_in[0];
  const int B = in_sizes[0] / 22;
  // inputs: 3:w1_44_s2 4:w1_66_s2 7:w1_44_t 8:w1_46_t 9:w1_64_t 10:w1_66_t 16:w2_tt
  ged_setup_kernel<<<1, 128, 0, stream>>>((const float*)d_in[3], (const float*)d_in[4],
                                          (const float*)d_in[7], (const float*)d_in[8],
                                          (const float*)d_in[9], (const float*)d_in[10],
                                          (const float*)d_in[16], d_ws);
  const int nblk = (B + 63) / 64;
  ged_main_kernel<<<nblk, 64, 0, stream>>>(feat, d_ws, (float*)d_out, B);
}
